// Multi_Head_Attention_20495583936565
// MI455X (gfx1250) — matrix operand order repaired and checked by reading, not compiled, not run on hardware
//
#include <hip/hip_runtime.h>

#define Bsz 2
#define Tt  2048
#define Cc  1024
#define Hh  16
#define Dd  64

static constexpr float kScale = 0.125f; // 1/sqrt(64), exact power of two

typedef __attribute__((ext_vector_type(16))) _Float16 v16h;
typedef __attribute__((ext_vector_type(8)))  _Float16 v8h;
typedef __attribute__((ext_vector_type(4)))  _Float16 v4h;
typedef __attribute__((ext_vector_type(8)))  float    v8f;
typedef __attribute__((ext_vector_type(4)))  float    v4f;

__device__ __forceinline__ v16h join16(v8h lo, v8h hi) {
  union { v16h v; v8h h[2]; } u;
  u.h[0] = lo; u.h[1] = hi;
  return u.v;
}

// ---- 16-lane butterfly reductions, one VOP2-DPP16 instruction per step ----
// DPP is an operand modifier: v_max_num_f32 dst, src0(dpp), src1 does
// shuffle+combine in a single instruction (vs mov_dpp + canonicalize + max
// that the compiler emits for the builtin path). Steps: xor1, xor2,
// row_half_mirror (=xor7 ~ xor4 once quads uniform), row_mirror (=xor15 ~
// xor8 once octets uniform); DPP rows are 16 lanes, so nothing crosses the
// 16-lane halves of the WMMA C-matrix layout.
__device__ __forceinline__ float rowmax16(float v) {
  asm("v_max_num_f32 %0, %1, %1 quad_perm:[1,0,3,2] row_mask:0xf bank_mask:0xf bound_ctrl:1"
      : "=v"(v) : "v"(v));
  asm("v_max_num_f32 %0, %1, %1 quad_perm:[2,3,0,1] row_mask:0xf bank_mask:0xf bound_ctrl:1"
      : "=v"(v) : "v"(v));
  asm("v_max_num_f32 %0, %1, %1 row_half_mirror row_mask:0xf bank_mask:0xf bound_ctrl:1"
      : "=v"(v) : "v"(v));
  asm("v_max_num_f32 %0, %1, %1 row_mirror row_mask:0xf bank_mask:0xf bound_ctrl:1"
      : "=v"(v) : "v"(v));
  return v;
}
__device__ __forceinline__ float rowsum16(float v) {
  asm("v_add_f32 %0, %1, %1 quad_perm:[1,0,3,2] row_mask:0xf bank_mask:0xf bound_ctrl:1"
      : "=v"(v) : "v"(v));
  asm("v_add_f32 %0, %1, %1 quad_perm:[2,3,0,1] row_mask:0xf bank_mask:0xf bound_ctrl:1"
      : "=v"(v) : "v"(v));
  asm("v_add_f32 %0, %1, %1 row_half_mirror row_mask:0xf bank_mask:0xf bound_ctrl:1"
      : "=v"(v) : "v"(v));
  asm("v_add_f32 %0, %1, %1 row_mirror row_mask:0xf bank_mask:0xf bound_ctrl:1"
      : "=v"(v) : "v"(v));
  return v;
}

// A-fragment 16x32 f16 (M x K). Arow points at row-block base; contiguous along K.
// Layout (ISA 7.12.2): lane<16: halves 0..7 -> K 0..7, halves 8..15 -> K 16..23
//                      lane>=16: halves 0..7 -> K 8..15, halves 8..15 -> K 24..31
__device__ __forceinline__ v16h load_a_frag(const _Float16* Arow, int lda, int kbase, int lane) {
  int m    = lane & 15;
  int koff = (lane < 16) ? 0 : 8;
  const _Float16* p = Arow + (size_t)m * lda + kbase + koff;
  v8h lo = *(const v8h*)p;
  v8h hi = *(const v8h*)(p + 16);
  return join16(lo, hi);
}

// B-fragment 32x16 f16 (K x N) for NT operands: row n of Bsrc is contiguous along K.
// Layout: lane<16 holds K 0..7,16..23 of column n=lane, lane>=16 holds K 8..15,24..31.
__device__ __forceinline__ v16h load_b_frag_nt(const _Float16* Bnrow, int ldb, int kbase, int lane) {
  int n    = lane & 15;
  int koff = (lane < 16) ? 0 : 8;
  const _Float16* p = Bnrow + (size_t)n * ldb + kbase + koff;
  v8h lo = *(const v8h*)p;
  v8h hi = *(const v8h*)(p + 16);
  return join16(lo, hi);
}

__device__ __forceinline__ v8f wmma_f16(v16h a, v16h b, v8f c) {
  return __builtin_amdgcn_wmma_f32_16x16x32_f16(false, a, false, b, (short)0, c, false, false);
}

// Vectorized f32 -> f16 convert (pure bandwidth; float4 in, 4xf16 out).
__global__ void cvt_f32_f16_x4(const float* __restrict__ in, _Float16* __restrict__ out, int n4) {
  int i = blockIdx.x * 256 + threadIdx.x;
  if (i < n4) {
    v4f f = *(const v4f*)(in + 4 * (size_t)i);
    v4h h;
#pragma unroll
    for (int j = 0; j < 4; ++j) h[j] = (_Float16)f[j];
    *(v4h*)(out + 4 * (size_t)i) = h;
  }
}

// Vh (B*T, C) row-major -> Vt (B, H, D, T): makes P*V B-fragments contiguous along K.
__global__ void transpose_v(const _Float16* __restrict__ Vh, _Float16* __restrict__ Vt) {
  size_t i = (size_t)blockIdx.x * 256 + threadIdx.x;
  const size_t total = (size_t)Bsz * Tt * Cc;
  if (i >= total) return;
  int    c  = (int)(i & (Cc - 1));
  size_t bt = i >> 10;                 // /C
  int    t  = (int)(bt & (Tt - 1));
  int    b  = (int)(bt >> 11);         // /T
  int    h  = c >> 6, d = c & 63;
  Vt[((size_t)(b * Hh + h) * Dd + d) * Tt + t] = Vh[i];
}

// C[M,N] = A[M,K] * W[N,K]^T, f16 inputs, f32 accumulate.
// 256 threads = 8 waves; wave w -> rows [blockIdx.y*256 + 32w, +32), cols [blockIdx.x*64, +64).
template <bool F32OUT>
__global__ __launch_bounds__(256) void gemm_nt(const _Float16* __restrict__ A,
                                               const _Float16* __restrict__ W,
                                               void* __restrict__ out,
                                               int M, int N, int K) {
  int lane  = threadIdx.x & 31;
  int wave  = threadIdx.x >> 5;
  int mrow  = blockIdx.y * 256 + wave * 32;
  int nbase = blockIdx.x * 64;
  const _Float16* Arow0 = A + (size_t)mrow * K;
  const _Float16* Arow1 = Arow0 + (size_t)16 * K;

  v8f acc[2][4] = {{{}, {}, {}, {}}, {{}, {}, {}, {}}};
  for (int k = 0; k < K; k += 32) {
    v16h a0 = load_a_frag(Arow0, K, k, lane);
    v16h a1 = load_a_frag(Arow1, K, k, lane);
#pragma unroll
    for (int j = 0; j < 4; ++j) {
      v16h bf = load_b_frag_nt(W + (size_t)(nbase + 16 * j) * K, K, k, lane);
      acc[0][j] = wmma_f16(a0, bf, acc[0][j]);
      acc[1][j] = wmma_f16(a1, bf, acc[1][j]);
    }
  }

  int n  = lane & 15;
  int mo = (lane < 16) ? 0 : 8;  // C/D layout: upper lanes hold M = 8 + r
#pragma unroll
  for (int g = 0; g < 2; ++g) {
#pragma unroll
    for (int j = 0; j < 4; ++j) {
#pragma unroll
      for (int r = 0; r < 8; ++r) {
        size_t idx = (size_t)(mrow + 16 * g + mo + r) * N + nbase + 16 * j + n;
        if (F32OUT) ((float*)out)[idx] = acc[g][j][r];
        else        ((_Float16*)out)[idx] = (_Float16)acc[g][j][r];
      }
    }
  }
}

// One 32-key flash-attention block for a 16-row query tile.
// MASKED=true only for the single diagonal block.
template <bool MASKED>
__device__ __forceinline__ void attn_block(int s0, int mrow, int lane, int nlane, int mo,
                                           const _Float16* __restrict__ Kb,
                                           const _Float16* __restrict__ Vb,
                                           _Float16* __restrict__ myp,
                                           v16h aq0, v16h aq1,
                                           v8f (&acc)[4], float (&mi)[8], float (&li)[8]) {
  if (!MASKED) {
    // Prefetch next key block's K rows and V stripe (global_prefetch_b8,
    // speculative: OOB prefetches are silently dropped).
    const _Float16* pk = Kb + (size_t)(s0 + 32 + nlane) * Cc + ((lane < 16) ? 0 : 32);
    __builtin_prefetch(pk, 0, 3);
    const _Float16* pv = Vb + (size_t)(nlane + ((lane < 16) ? 0 : 16)) * Tt + s0 + 32;
    __builtin_prefetch(pv, 0, 3);
  }

  // S = Q K^T for 16 queries x 32 keys (two 16x16 tiles, K-dim = 64)
  v8f st0 = {}, st1 = {};
  {
    v16h b00 = load_b_frag_nt(Kb + (size_t)s0 * Cc,        Cc, 0,  lane);
    v16h b01 = load_b_frag_nt(Kb + (size_t)s0 * Cc,        Cc, 32, lane);
    st0 = wmma_f16(aq0, b00, st0);
    st0 = wmma_f16(aq1, b01, st0);
    v16h b10 = load_b_frag_nt(Kb + (size_t)(s0 + 16) * Cc, Cc, 0,  lane);
    v16h b11 = load_b_frag_nt(Kb + (size_t)(s0 + 16) * Cc, Cc, 32, lane);
    st1 = wmma_f16(aq0, b10, st1);
    st1 = wmma_f16(aq1, b11, st1);
  }

  // online softmax (row = mo + r, cols across 16 lanes); reductions via DPP
#pragma unroll
  for (int r = 0; r < 8; ++r) {
    float a0 = st0[r];
    float a1 = st1[r];
    if (MASKED) {
      int mg = mrow + mo + r;
      if (s0 + nlane > mg)      a0 = -1e30f;
      if (s0 + 16 + nlane > mg) a1 = -1e30f;
    }
    float bm    = rowmax16(fmaxf(a0, a1));
    float mnew  = fmaxf(mi[r], bm);
    float alpha = __expf(mi[r] - mnew);
    float p0 = __expf(a0 - mnew);
    float p1 = __expf(a1 - mnew);
    float rs = rowsum16(p0 + p1);
    li[r] = li[r] * alpha + rs;
    mi[r] = mnew;
#pragma unroll
    for (int d = 0; d < 4; ++d) acc[d][r] *= alpha;
    int mloc = mo + r;
    myp[mloc * 32 + nlane]      = (_Float16)p0;
    myp[mloc * 32 + nlane + 16] = (_Float16)p1;
  }
  asm volatile("s_wait_dscnt 0" ::: "memory");

  // Reload P (16x32) in A-fragment swizzle from per-wave LDS
  v16h ap;
  {
    int m    = lane & 15;
    int koff = (lane < 16) ? 0 : 8;
    const _Float16* p = myp + m * 32 + koff;
    v8h lo = *(const v8h*)p;
    v8h hi = *(const v8h*)(p + 16);
    ap = join16(lo, hi);
  }
  // O += P * V  (V transposed: rows = d, contiguous along keys)
#pragma unroll
  for (int d = 0; d < 4; ++d) {
    v16h bv = load_b_frag_nt(Vb + (size_t)(16 * d) * Tt, Tt, s0, lane);
    acc[d] = wmma_f16(ap, bv, acc[d]);
  }
}

// Causal flash attention. 1 block = 128 query rows (8 waves x 16); online softmax.
__global__ __launch_bounds__(256) void flash_attn(const _Float16* __restrict__ Qh,
                                                  const _Float16* __restrict__ Kh,
                                                  const _Float16* __restrict__ Vt,
                                                  _Float16* __restrict__ Oh) {
  __shared__ _Float16 psh[8][16 * 32];  // per-wave P tile staging (8 KB)

  int lane = threadIdx.x & 31;
  int wave = threadIdx.x >> 5;
  int qblk = blockIdx.x & 15;           // T/128 = 16
  int hb   = blockIdx.x >> 4;
  int h    = hb & (Hh - 1);
  int b    = hb >> 4;
  int mrow = qblk * 128 + wave * 16;

  const _Float16* Qb = Qh + ((size_t)b * Tt) * Cc + h * Dd;
  const _Float16* Kb = Kh + ((size_t)b * Tt) * Cc + h * Dd;
  const _Float16* Vb = Vt + ((size_t)(b * Hh + h) * Dd) * Tt;

  v16h aq0 = load_a_frag(Qb + (size_t)mrow * Cc, Cc, 0,  lane);
  v16h aq1 = load_a_frag(Qb + (size_t)mrow * Cc, Cc, 32, lane);
  // Pre-scale Q by 1/sqrt(D)=2^-3: exact in f16 (exponent shift only).
#pragma unroll
  for (int i = 0; i < 16; ++i) { aq0[i] *= (_Float16)kScale; aq1[i] *= (_Float16)kScale; }

  v8f acc[4] = {{}, {}, {}, {}};
  float mi[8], li[8];
#pragma unroll
  for (int r = 0; r < 8; ++r) { mi[r] = -1e30f; li[r] = 0.f; }

  int nlane = lane & 15;
  int mo    = (lane < 16) ? 0 : 8;
  _Float16* myp = &psh[wave][0];

  // Steady state: blocks fully below the diagonal (s0+31 <= mrow) need no mask.
  int s0 = 0;
  for (; s0 + 31 <= mrow; s0 += 32)
    attn_block<false>(s0, mrow, lane, nlane, mo, Kb, Vb, myp, aq0, aq1, acc, mi, li);
  // Exactly one diagonal block remains (keys [s0, s0+31] straddle rows [mrow, mrow+15]).
  attn_block<true>(s0, mrow, lane, nlane, mo, Kb, Vb, myp, aq0, aq1, acc, mi, li);

#pragma unroll
  for (int d = 0; d < 4; ++d) {
#pragma unroll
    for (int r = 0; r < 8; ++r) {
      float o = acc[d][r] / li[r];
      Oh[((size_t)b * Tt + mrow + mo + r) * Cc + h * Dd + 16 * d + nlane] = (_Float16)o;
    }
  }
}

extern "C" void kernel_launch(void* const* d_in, const int* in_sizes, int n_in,
                              void* d_out, int out_size, void* d_ws, size_t ws_size,
                              hipStream_t stream) {
  const float* x  = (const float*)d_in[0];
  const float* Wq = (const float*)d_in[1];
  const float* Wk = (const float*)d_in[2];
  const float* Wv = (const float*)d_in[3];
  const float* Wo = (const float*)d_in[4];

  const int M    = Bsz * Tt;   // 4096
  const int NE_X = M * Cc;     // 4,194,304
  const int NE_W = Cc * Cc;    // 1,048,576

  _Float16* ws  = (_Float16*)d_ws;
  _Float16* xh  = ws;
  _Float16* wqh = xh  + NE_X;
  _Float16* wkh = wqh + NE_W;
  _Float16* wvh = wkh + NE_W;
  _Float16* woh = wvh + NE_W;
  _Float16* qh  = woh + NE_W;
  _Float16* kh  = qh  + NE_X;
  _Float16* vh  = kh  + NE_X;
  _Float16* vt  = vh  + NE_X;
  _Float16* oh  = vt  + NE_X;   // total ~59 MB of workspace

  cvt_f32_f16_x4<<<(NE_X / 4 + 255) / 256, 256, 0, stream>>>(x,  xh,  NE_X / 4);
  cvt_f32_f16_x4<<<(NE_W / 4 + 255) / 256, 256, 0, stream>>>(Wq, wqh, NE_W / 4);
  cvt_f32_f16_x4<<<(NE_W / 4 + 255) / 256, 256, 0, stream>>>(Wk, wkh, NE_W / 4);
  cvt_f32_f16_x4<<<(NE_W / 4 + 255) / 256, 256, 0, stream>>>(Wv, wvh, NE_W / 4);
  cvt_f32_f16_x4<<<(NE_W / 4 + 255) / 256, 256, 0, stream>>>(Wo, woh, NE_W / 4);

  dim3 ggrid(Cc / 64, M / 256);  // (16, 16)
  gemm_nt<false><<<ggrid, 256, 0, stream>>>(xh, wqh, qh, M, Cc, Cc);
  gemm_nt<false><<<ggrid, 256, 0, stream>>>(xh, wkh, kh, M, Cc, Cc);
  gemm_nt<false><<<ggrid, 256, 0, stream>>>(xh, wvh, vh, M, Cc, Cc);

  transpose_v<<<(NE_X + 255) / 256, 256, 0, stream>>>(vh, vt);

  flash_attn<<<Bsz * Hh * (Tt / 128), 256, 0, stream>>>(qh, kh, vt, oh);

  gemm_nt<true><<<ggrid, 256, 0, stream>>>(oh, woh, d_out, M, Cc, Cc);
}
